// FieldFormer_TAP_81578608820821
// MI455X (gfx1250) — compile-verified, matrix operand order checked by reading, not run on hardware
//
#include <hip/hip_runtime.h>
#include <hip/hip_bf16.h>
#include <math.h>
#include <stdint.h>

// ---------------- problem constants ----------------
#define PNUM   6859          // patches (19^3)
#define PPAD   6912          // padded: multiple of 128 (block tile M and N)
#define DIMF   512           // features per patch
#define FM     361           // 19*19
#define NAXI   19
#define RANK   80
#define SCORE_SCALE 0.04419417382415922f   // 1/sqrt(512)

// LDS tile geometry: 128 rows x 32 K, row stride padded to 40 ushorts (80 B = 20 banks,
// gcd(20,64)=4 -> 16 distinct bank offsets across 16 rows; all frag offsets stay 16B-aligned)
#define TROWS  128
#define TK     32
#define TSTR   40

typedef __attribute__((ext_vector_type(16))) __bf16 bf16x16;
typedef __attribute__((ext_vector_type(8)))  float  floatx8;
typedef int v4i __attribute__((vector_size(16)));   // builtin's expected pointee type

struct Frag128x2 { uint4 lo, hi; };

// ---- optional CDNA5 async global->LDS path (probed at compile time) ----
#if defined(__has_builtin)
#  if __has_builtin(__builtin_amdgcn_global_load_async_to_lds_b128) && \
      __has_builtin(__builtin_amdgcn_s_wait_asynccnt)
#    define USE_ASYNC_LDS 1
#  endif
#endif

// copy 16 bytes global -> LDS (per lane)
__device__ __forceinline__ void g2l16(const unsigned short* g, unsigned short* l) {
#ifdef USE_ASYNC_LDS
  __builtin_amdgcn_global_load_async_to_lds_b128((v4i*)g, (v4i*)l, 0, 0);
#else
  *(uint4*)l = *(const uint4*)g;
#endif
}

__device__ __forceinline__ void sync_copies() {
#ifdef USE_ASYNC_LDS
  __builtin_amdgcn_s_wait_asynccnt(0);
#endif
  __syncthreads();
}

// ---------------- helpers ----------------
__device__ __forceinline__ unsigned short f2bf(float f) {
  unsigned int u = __float_as_uint(f);
  unsigned int r = u + 0x7FFFu + ((u >> 16) & 1u);   // round-to-nearest-even
  return (unsigned short)(r >> 16);
}

// Load one bf16 WMMA fragment for this lane from a K-major row at base p
// (p already includes row*stride + kb, kb = (lane>>4)*8): K = kb+[0..7] and kb+16+[0..7].
__device__ __forceinline__ bf16x16 ldfrag(const unsigned short* p) {
  Frag128x2 f;
  f.lo = *(const uint4*)(p);
  f.hi = *(const uint4*)(p + 16);
  return __builtin_bit_cast(bf16x16, f);
}

__device__ __forceinline__ void store_tile(float* C, int ldc, int m0, int n0, int lane,
                                           const floatx8& acc, float alpha,
                                           int mMax, int nMax) {
  int col   = n0 + (lane & 15);
  int rbase = m0 + ((lane >> 4) << 3);     // VGPR r -> row rbase + r
  if (col < nMax) {
#pragma unroll
    for (int r = 0; r < 8; ++r) {
      int row = rbase + r;
      if (row < mMax) C[(size_t)row * ldc + col] = alpha * acc[r];
    }
  }
}

__device__ __forceinline__ float block_reduce_sum(float v, float* red) {
  int t = threadIdx.x;
  red[t] = v; __syncthreads();
  for (int o = 128; o > 0; o >>= 1) { if (t < o) red[t] += red[t + o]; __syncthreads(); }
  float r = red[0]; __syncthreads();
  return r;
}

__device__ __forceinline__ float block_reduce_max(float v, float* red) {
  int t = threadIdx.x;
  red[t] = v; __syncthreads();
  for (int o = 128; o > 0; o >>= 1) { if (t < o) red[t] = fmaxf(red[t], red[t + o]); __syncthreads(); }
  float r = red[0]; __syncthreads();
  return r;
}

// ---------------- kernels ----------------

// f32 -> bf16 pack with row padding (rows >= validRows written as 0)
__global__ void pack_bf16_kernel(const float* __restrict__ src, unsigned short* __restrict__ dst,
                                 int rows, int cols, int validRows) {
  long i = (long)blockIdx.x * blockDim.x + threadIdx.x;
  long total = (long)rows * cols;
  if (i >= total) return;
  int r = (int)(i / cols);
  dst[i] = (r < validRows) ? f2bf(src[i]) : (unsigned short)0;
}

// Stage one 128x32 bf16 tile (row-major, ld = K) into LDS (stride TSTR).
// 512 x 16B chunks over 256 threads (2 chunks each).
__device__ __forceinline__ void copy_tile(const unsigned short* __restrict__ g, int ldg,
                                          unsigned short* l) {
  const int t = threadIdx.x;
#pragma unroll
  for (int c = 0; c < 2; ++c) {
    int chunk = t + c * 256;
    int row = chunk >> 2;          // 0..127
    int seg = chunk & 3;           // 0..3 (8 ushorts each)
    const unsigned short* gp = g + (size_t)row * ldg + seg * 8;
    __builtin_prefetch(gp + 2 * TK, 0, 1);   // global_prefetch_b8, two K-tiles ahead
    g2l16(gp, l + row * TSTR + seg * 8);
  }
}

// C[M x N] = alpha * A[M x K] * B[N x K]^T, bf16 in, f32 out.
// Block tile 128x128, double-buffered LDS staging, K-step 32.
// 8 waves as 2(M) x 4(N); wave tile 64x32 -> 8 accumulators (4 A-frags x 2 B-frags).
__global__ __launch_bounds__(256) void gemm_bf16_wmma(const unsigned short* __restrict__ A,
                                                      const unsigned short* __restrict__ B,
                                                      float* __restrict__ C,
                                                      int K, int ldc, float alpha,
                                                      int mMax, int nMax) {
  __shared__ __align__(16) unsigned short As[2][TROWS * TSTR];
  __shared__ __align__(16) unsigned short Bs[2][TROWS * TSTR];

  const int lane = threadIdx.x & 31;
  const int wave = threadIdx.x >> 5;
  const int ra = lane & 15;
  const int kb = (lane >> 4) * 8;
  const int mw = (wave >> 2) * 64;     // wave row offset in block tile
  const int nw = (wave & 3) * 32;      // wave col offset in block tile

  const unsigned short* Ab = A + (size_t)(blockIdx.y * TROWS) * K;
  const unsigned short* Bb = B + (size_t)(blockIdx.x * TROWS) * K;
  const int nK = K >> 5;

  floatx8 acc[4][2] = {};

  copy_tile(Ab, K, As[0]);
  copy_tile(Bb, K, Bs[0]);
  sync_copies();

  for (int kt = 0; kt < nK; ++kt) {
    const int cur = kt & 1;
    if (kt + 1 < nK) {                         // prefetch next tile into the other buffer
      copy_tile(Ab + (kt + 1) * TK, K, As[cur ^ 1]);
      copy_tile(Bb + (kt + 1) * TK, K, Bs[cur ^ 1]);
    }

    bf16x16 a[4], b[2];
#pragma unroll
    for (int f = 0; f < 4; ++f)
      a[f] = ldfrag(&As[cur][(mw + f * 16 + ra) * TSTR + kb]);
#pragma unroll
    for (int t2 = 0; t2 < 2; ++t2)
      b[t2] = ldfrag(&Bs[cur][(nw + t2 * 16 + ra) * TSTR + kb]);

#pragma unroll
    for (int f = 0; f < 4; ++f)
#pragma unroll
      for (int t2 = 0; t2 < 2; ++t2)
        acc[f][t2] = __builtin_amdgcn_wmma_f32_16x16x32_bf16(
            false, a[f], false, b[t2], (short)0, acc[f][t2], false, false);

    sync_copies();   // own async copies landed + all waves done reading 'cur'
  }

  const int m0 = blockIdx.y * TROWS + mw;
  const int n0 = blockIdx.x * TROWS + nw;
#pragma unroll
  for (int f = 0; f < 4; ++f)
#pragma unroll
    for (int t2 = 0; t2 < 2; ++t2)
      store_tile(C, ldc, m0 + f * 16, n0 + t2 * 16, lane, acc[f][t2], alpha, mMax, nMax);
}

// Per-row BatchNorm (train stats over 512 features) + affine + L2 norm -> bf16 rows;
// pad rows [validRows, PPAD) zeroed. One 256-thread block per row.
__global__ __launch_bounds__(256) void bn_l2_bf16_kernel(const float* __restrict__ X,
                                                         const float* __restrict__ gamma,
                                                         const float* __restrict__ beta,
                                                         unsigned short* __restrict__ Y,
                                                         int validRows) {
  __shared__ float red[256];
  const int row = blockIdx.x;
  const int t = threadIdx.x;
  unsigned short* yr = Y + (size_t)row * DIMF;
  if (row >= validRows) {
    yr[t] = 0; yr[t + 256] = 0;
    return;
  }
  const float* x = X + (size_t)row * DIMF;
  float x0 = x[t], x1 = x[t + 256];

  float mu  = block_reduce_sum(x0 + x1, red) * (1.0f / DIMF);
  float d0 = x0 - mu, d1 = x1 - mu;
  float var = block_reduce_sum(d0 * d0 + d1 * d1, red) * (1.0f / DIMF);
  float inv = rsqrtf(var + 1e-5f);
  float g = gamma[row], b = beta[row];
  float y0 = fmaf(g, d0 * inv, b);
  float y1 = fmaf(g, d1 * inv, b);
  float ss = block_reduce_sum(y0 * y0 + y1 * y1, red);
  float rn = 1.0f / fmaxf(sqrtf(ss), 1e-12f);
  yr[t]       = f2bf(y0 * rn);
  yr[t + 256] = f2bf(y1 * rn);
}

// Sparsemax per row, in place: tau solves sum(max(z-tau,0))=1, bisected on [zmax-1, zmax].
__global__ __launch_bounds__(256) void sparsemax_kernel(float* __restrict__ att) {
  __shared__ float z[PNUM];
  __shared__ float red[256];
  const int t = threadIdx.x;
  float* rp = att + (size_t)blockIdx.x * PNUM;

  float lmax = -3.4e38f;
  for (int i = t; i < PNUM; i += 256) { float v = rp[i]; z[i] = v; lmax = fmaxf(lmax, v); }
  float zmax = block_reduce_max(lmax, red);

  float lo = zmax - 1.0f, hi = zmax;
  for (int it = 0; it < 32; ++it) {
    float tau = 0.5f * (lo + hi);
    float ls = 0.0f;
    for (int i = t; i < PNUM; i += 256) ls += fmaxf(z[i] - tau, 0.0f);
    float s = block_reduce_sum(ls, red);
    if (s >= 1.0f) lo = tau; else hi = tau;   // uniform across block: no divergence
  }
  float tau = 0.5f * (lo + hi);
  for (int i = t; i < PNUM; i += 256) rp[i] = fmaxf(z[i] - tau, 0.0f);
}

// Mode-1 Tucker contraction fused with the 6D permute gather:
// T1[i,y,z] = sum_x U1[i,x] * att[n(x1,y1,z1), m(x2,y2,z2)]
__global__ __launch_bounds__(128) void tucker_mode1(const float* __restrict__ att,
                                                    const float* __restrict__ U1,
                                                    float* __restrict__ T1) {
  __shared__ float v[FM];
  const int yz = blockIdx.x;            // [0, FM*FM)
  const int y = yz / FM, z = yz % FM;
  const int y1 = y / NAXI, y2 = y % NAXI;
  const int z1 = z / NAXI, z2 = z % NAXI;
  for (int x = threadIdx.x; x < FM; x += blockDim.x) {
    int x1 = x / NAXI, x2 = x % NAXI;
    int n = x1 * FM + y1 * NAXI + z1;
    int m = x2 * FM + y2 * NAXI + z2;
    v[x] = att[(size_t)n * PNUM + m];
  }
  __syncthreads();
  const int i = threadIdx.x;
  if (i < RANK) {
    float s = 0.0f;
    const float* u = U1 + i * FM;
#pragma unroll 4
    for (int x = 0; x < FM; ++x) s = fmaf(u[x], v[x], s);
    T1[((size_t)i * FM + y) * FM + z] = s;
  }
}

// T2[i,j,z] = sum_y U2[j,y] * T1[i,y,z]
__global__ __launch_bounds__(128) void tucker_mode2(const float* __restrict__ T1,
                                                    const float* __restrict__ U2,
                                                    float* __restrict__ T2) {
  __shared__ float v[FM];
  const int iz = blockIdx.x;            // [0, RANK*FM)
  const int i = iz / FM, z = iz % FM;
  for (int y = threadIdx.x; y < FM; y += blockDim.x)
    v[y] = T1[((size_t)i * FM + y) * FM + z];
  __syncthreads();
  const int j = threadIdx.x;
  if (j < RANK) {
    float s = 0.0f;
    const float* u = U2 + j * FM;
#pragma unroll 4
    for (int y = 0; y < FM; ++y) s = fmaf(u[y], v[y], s);
    T2[((size_t)(i * RANK + j)) * FM + z] = s;
  }
}

// Y[i,j,k] = tanh( sum_z U3[k,z] * T2[i,j,z] )
__global__ __launch_bounds__(128) void tucker_mode3(const float* __restrict__ T2,
                                                    const float* __restrict__ U3,
                                                    float* __restrict__ Y) {
  __shared__ float v[FM];
  const int ij = blockIdx.x;            // [0, RANK*RANK)
  for (int z = threadIdx.x; z < FM; z += blockDim.x)
    v[z] = T2[(size_t)ij * FM + z];
  __syncthreads();
  const int k = threadIdx.x;
  if (k < RANK) {
    float s = 0.0f;
    const float* u = U3 + k * FM;
#pragma unroll 4
    for (int z = 0; z < FM; ++z) s = fmaf(u[z], v[z], s);
    Y[(size_t)ij * RANK + k] = tanhf(s);
  }
}

// ---------------- launch ----------------
extern "C" void kernel_launch(void* const* d_in, const int* in_sizes, int n_in,
                              void* d_out, int out_size, void* d_ws, size_t ws_size,
                              hipStream_t stream) {
  (void)in_sizes; (void)n_in; (void)out_size; (void)ws_size;

  const float* patches = (const float*)d_in[0];   // (P, 512)
  const float* Wq      = (const float*)d_in[1];   // (512, 512), row d, col c
  const float* Wk      = (const float*)d_in[2];
  const float* gamma   = (const float*)d_in[3];   // (P,)
  const float* beta    = (const float*)d_in[4];
  const float* U1      = (const float*)d_in[5];   // (80, 361)
  const float* U2      = (const float*)d_in[6];
  const float* U3      = (const float*)d_in[7];

  float* y_out   = (float*)d_out;                      // (80,80,80)
  float* att_out = y_out + (size_t)RANK * RANK * RANK; // (P, P)

  // workspace layout
  char* ws = (char*)d_ws;
  unsigned short* Xbf  = (unsigned short*)ws;                 ws += (size_t)PPAD * DIMF * 2;
  unsigned short* Wqbf = (unsigned short*)ws;                 ws += (size_t)DIMF * DIMF * 2;
  unsigned short* Wkbf = (unsigned short*)ws;                 ws += (size_t)DIMF * DIMF * 2;
  unsigned short* Qbf  = (unsigned short*)ws;                 ws += (size_t)PPAD * DIMF * 2;
  unsigned short* Kbf  = (unsigned short*)ws;                 ws += (size_t)PPAD * DIMF * 2;
  float*          T1   = (float*)ws;                          ws += (size_t)RANK * FM * FM * 4;
  float*          T2   = (float*)ws;                          ws += (size_t)RANK * RANK * FM * 4;
  // raw (pre-BN) projections live temporarily in the (much larger) att region of d_out
  float* Qraw = att_out;
  float* Kraw = att_out + (size_t)PPAD * DIMF;

  // 1) pack inputs to bf16 (X rows padded to PPAD with zeros)
  {
    long tot = (long)PPAD * DIMF;
    pack_bf16_kernel<<<(tot + 255) / 256, 256, 0, stream>>>(patches, Xbf, PPAD, DIMF, PNUM);
    long wt = (long)DIMF * DIMF;
    pack_bf16_kernel<<<(wt + 255) / 256, 256, 0, stream>>>(Wq, Wqbf, DIMF, DIMF, DIMF);
    pack_bf16_kernel<<<(wt + 255) / 256, 256, 0, stream>>>(Wk, Wkbf, DIMF, DIMF, DIMF);
  }

  // 2) projections: Qraw = X * Wq^T  (M=PPAD, N=512, K=512)
  {
    dim3 grid(DIMF / 128, PPAD / 128);
    gemm_bf16_wmma<<<grid, 256, 0, stream>>>(Xbf, Wqbf, Qraw, DIMF, DIMF, 1.0f, PPAD, DIMF);
    gemm_bf16_wmma<<<grid, 256, 0, stream>>>(Xbf, Wkbf, Kraw, DIMF, DIMF, 1.0f, PPAD, DIMF);
  }

  // 3) BN (train stats) + affine + L2 normalize -> bf16 rows (pad rows zeroed)
  bn_l2_bf16_kernel<<<PPAD, 256, 0, stream>>>(Qraw, gamma, beta, Qbf, PNUM);
  bn_l2_bf16_kernel<<<PPAD, 256, 0, stream>>>(Kraw, gamma, beta, Kbf, PNUM);

  // 4) scores = (Q * K^T) / sqrt(512)  (M=N=PPAD, K=512) -> att_out
  {
    dim3 grid(PPAD / 128, PPAD / 128);
    gemm_bf16_wmma<<<grid, 256, 0, stream>>>(Qbf, Kbf, att_out, DIMF, PNUM, SCORE_SCALE, PNUM, PNUM);
  }

  // 5) sparsemax per row, in place
  sparsemax_kernel<<<PNUM, 256, 0, stream>>>(att_out);

  // 6) permute-fused Tucker contractions + tanh
  tucker_mode1<<<FM * FM, 128, 0, stream>>>(att_out, U1, T1);
  tucker_mode2<<<RANK * FM, 128, 0, stream>>>(T1, U2, T2);
  tucker_mode3<<<RANK * RANK, 128, 0, stream>>>(T2, U3, y_out);
}
